// GatedDeltaNetCell_4887672783151
// MI455X (gfx1250) — matrix operand order repaired and checked by reading, not compiled, not run on hardware
//
#include <hip/hip_runtime.h>
#include <math.h>

#define T_DIM 1024
#define B_DIM 8
#define D_DIM 1024
#define H_DIM 16
#define DH_DIM 64
#define N_TOK (T_DIM * B_DIM) /* 8192 */

typedef __attribute__((ext_vector_type(16))) __bf16 v16bf;
typedef __attribute__((ext_vector_type(8))) float v8f;

__device__ __forceinline__ unsigned short f2bf(float f) {
    unsigned u = __float_as_uint(f);
    u += 0x7FFFu + ((u >> 16) & 1u); // round-to-nearest-even
    return (unsigned short)(u >> 16);
}

// ---------------------------------------------------------------------------
// f32 -> bf16 conversion (4 elements / thread)
// ---------------------------------------------------------------------------
__global__ __launch_bounds__(256) void convert_f32_bf16(const float* __restrict__ src,
                                                        unsigned short* __restrict__ dst,
                                                        int n) {
    int i = (blockIdx.x * 256 + threadIdx.x) * 4;
    if (i + 3 < n) {
        float4 f = *(const float4*)(src + i);
        unsigned short t[4];
        t[0] = f2bf(f.x); t[1] = f2bf(f.y); t[2] = f2bf(f.z); t[3] = f2bf(f.w);
        *(uint2*)(dst + i) = *(const uint2*)t;
    }
}

// ---------------------------------------------------------------------------
// Projection GEMM: out[n, o] = sum_d x[n,d] * W[o,d]   (i.e. x @ W.T)
// block = 256 threads (8 waves). Each wave computes a 32x64 output tile:
// 2 M-tiles x 4 N-tiles = 8 accumulators of v_wmma_f32_16x16x32_bf16.
// All 6 fragments of a K-step (A0,A1,B0..B3) load into DISTINCT registers
// before any WMMA so the loads stay in flight across the 8 WMMAs (staged
// s_wait_loadcnt instead of full drains).
//   mode 0: raw write (v)
//   mode 1: L2-normalize each 64-wide row segment (q, k)
//   mode 2: sigmoid(x+bias), mean over 64 -> out is (N_TOK, H) (alpha, beta)
// ---------------------------------------------------------------------------
__global__ __launch_bounds__(256) void proj_gemm(const unsigned short* __restrict__ xb,
                                                 const unsigned short* __restrict__ wb,
                                                 const float* __restrict__ bias,
                                                 float* __restrict__ out,
                                                 int mode) {
    const int tid  = threadIdx.x;
    const int wave = tid >> 5;
    const int lane = tid & 31;
    const int lo   = lane & 15;
    const int hi   = lane >> 4;
    const int n0   = blockIdx.x * 256 + wave * 32; // 32-row tile per wave
    const int o0   = blockIdx.y * 64;              // col strip == head*64

    v8f acc[2][4];
#pragma unroll
    for (int m = 0; m < 2; ++m)
#pragma unroll
        for (int i = 0; i < 4; ++i)
#pragma unroll
            for (int j = 0; j < 8; ++j) acc[m][i][j] = 0.0f;

    // A-fragment K pair base offsets: contiguous k = 16*hi + e (matches B)
    const int pairK[8] = {0, 2, 4, 6, 8, 10, 12, 14};
    const unsigned short* xrow0 = xb + (size_t)(n0 + lo) * D_DIM + 16 * hi;
    const unsigned short* xrow1 = xrow0 + (size_t)16 * D_DIM;
    const unsigned short* wrow  = wb + (size_t)(o0 + lo) * D_DIM + 16 * hi;

    union AB { v16bf v; unsigned u[8]; uint4 q[2]; };

    for (int kb = 0; kb < D_DIM; kb += 32) {
        AB A0, A1, B0, B1, B2, B3;
#pragma unroll
        for (int p = 0; p < 8; ++p) {
            A0.u[p] = *(const unsigned*)(xrow0 + kb + pairK[p]);
            A1.u[p] = *(const unsigned*)(xrow1 + kb + pairK[p]);
        }
        // B[k, n] = W[o0+i*16+lo, kb + e + 16*hi] : 16 contiguous bf16
        B0.q[0] = *(const uint4*)(wrow + kb);
        B0.q[1] = *(const uint4*)(wrow + kb + 8);
        B1.q[0] = *(const uint4*)(wrow + (size_t)16 * D_DIM + kb);
        B1.q[1] = *(const uint4*)(wrow + (size_t)16 * D_DIM + kb + 8);
        B2.q[0] = *(const uint4*)(wrow + (size_t)32 * D_DIM + kb);
        B2.q[1] = *(const uint4*)(wrow + (size_t)32 * D_DIM + kb + 8);
        B3.q[0] = *(const uint4*)(wrow + (size_t)48 * D_DIM + kb);
        B3.q[1] = *(const uint4*)(wrow + (size_t)48 * D_DIM + kb + 8);

        acc[0][0] = __builtin_amdgcn_wmma_f32_16x16x32_bf16(
            false, A0.v, false, B0.v, (short)0, acc[0][0], false, false);
        acc[1][0] = __builtin_amdgcn_wmma_f32_16x16x32_bf16(
            false, A1.v, false, B0.v, (short)0, acc[1][0], false, false);
        acc[0][1] = __builtin_amdgcn_wmma_f32_16x16x32_bf16(
            false, A0.v, false, B1.v, (short)0, acc[0][1], false, false);
        acc[1][1] = __builtin_amdgcn_wmma_f32_16x16x32_bf16(
            false, A1.v, false, B1.v, (short)0, acc[1][1], false, false);
        acc[0][2] = __builtin_amdgcn_wmma_f32_16x16x32_bf16(
            false, A0.v, false, B2.v, (short)0, acc[0][2], false, false);
        acc[1][2] = __builtin_amdgcn_wmma_f32_16x16x32_bf16(
            false, A1.v, false, B2.v, (short)0, acc[1][2], false, false);
        acc[0][3] = __builtin_amdgcn_wmma_f32_16x16x32_bf16(
            false, A0.v, false, B3.v, (short)0, acc[0][3], false, false);
        acc[1][3] = __builtin_amdgcn_wmma_f32_16x16x32_bf16(
            false, A1.v, false, B3.v, (short)0, acc[1][3], false, false);
    }

    if (mode == 2) {
        // sigmoid + mean over the 64-wide head segment
        float bia[4];
#pragma unroll
        for (int i = 0; i < 4; ++i) bia[i] = bias[o0 + i * 16 + lo];
#pragma unroll
        for (int m = 0; m < 2; ++m)
#pragma unroll
            for (int j = 0; j < 8; ++j) {
                float s = 0.f;
#pragma unroll
                for (int i = 0; i < 4; ++i) {
                    float z = acc[m][i][j] + bia[i];
                    s += 1.0f / (1.0f + __expf(-z));
                }
                // reduce across the 16 lanes of this half-group (row j + 8*hi)
                s += __shfl_xor(s, 1); s += __shfl_xor(s, 2);
                s += __shfl_xor(s, 4); s += __shfl_xor(s, 8);
                if (lo == 0)
                    out[(size_t)(n0 + m * 16 + j + 8 * hi) * H_DIM + blockIdx.y] =
                        s * (1.0f / 64.0f);
            }
        return;
    }

    if (mode == 1) {
        // L2 norm over the 64-wide head segment per row
#pragma unroll
        for (int m = 0; m < 2; ++m)
#pragma unroll
            for (int j = 0; j < 8; ++j) {
                float s = 0.f;
#pragma unroll
                for (int i = 0; i < 4; ++i) s += acc[m][i][j] * acc[m][i][j];
                s += __shfl_xor(s, 1); s += __shfl_xor(s, 2);
                s += __shfl_xor(s, 4); s += __shfl_xor(s, 8);
                float inv = 1.0f / fmaxf(sqrtf(s), 1e-12f);
#pragma unroll
                for (int i = 0; i < 4; ++i) acc[m][i][j] *= inv;
            }
    }

#pragma unroll
    for (int m = 0; m < 2; ++m)
#pragma unroll
        for (int i = 0; i < 4; ++i)
#pragma unroll
            for (int j = 0; j < 8; ++j)
                out[(size_t)(n0 + m * 16 + j + 8 * hi) * D_DIM + o0 + i * 16 + lo] =
                    acc[m][i][j];
}

// ---------------------------------------------------------------------------
// Sequential gated delta scan, software-pipelined.
// One block per (b,h); S (64x64) lives in registers: thread (r=tid/4, c=tid%4)
// owns S[r, c*16 .. c*16+15]. LDS only for k/q/v broadcast + 4-way row sums.
// Step t+1's q/k/v/alpha/beta are fetched into registers while step t
// computes, hiding global latency behind the recurrence.
// ---------------------------------------------------------------------------
__global__ __launch_bounds__(256) void scan_kernel(const float* __restrict__ qn,
                                                   const float* __restrict__ kn,
                                                   const float* __restrict__ vn,
                                                   const float* __restrict__ ar,
                                                   const float* __restrict__ br,
                                                   const float* __restrict__ S0,
                                                   float* __restrict__ outs,
                                                   float* __restrict__ Sfin) {
    const int bh = blockIdx.x; // 0..127
    const int b = bh >> 4, h = bh & 15;
    const int tid = threadIdx.x;
    const int r = tid >> 2, c = tid & 3;
    const int pb = tid & ~3;

    __shared__ float qb[64], kb[64], vb[64], part[256];

    float s[16];
    const size_t sbase = ((size_t)bh * 64 + r) * 64 + c * 16;
#pragma unroll
    for (int j = 0; j < 16; ++j) s[j] = S0[sbase + j];

    // prologue: fetch t=0 operands into registers
    float ql = 0.f, kl = 0.f, vl = 0.f;
    {
        size_t base0 = (size_t)b * D_DIM + h * 64 + tid;
        if (tid < 64) { ql = qn[base0]; kl = kn[base0]; vl = vn[base0]; }
    }
    float a_next  = ar[(size_t)b * H_DIM + h];
    float be_next = br[(size_t)b * H_DIM + h];

    for (int t = 0; t < T_DIM; ++t) {
        const size_t n = (size_t)t * B_DIM + b;
        if (tid < 64) { qb[tid] = ql; kb[tid] = kl; vb[tid] = vl; }
        const float a = a_next, be = be_next;
        __syncthreads();

        // issue next step's loads; results land next iteration
        if (t + 1 < T_DIM) {
            const size_t n1 = n + B_DIM;
            if (tid < 64) {
                size_t base = n1 * D_DIM + h * 64 + tid;
                ql = qn[base]; kl = kn[base]; vl = vn[base];
            }
            a_next = ar[n1 * H_DIM + h];
            be_next = br[n1 * H_DIM + h];
        }

        // Sk = S @ k : partial dot over owned 16 columns
        float p = 0.f;
#pragma unroll
        for (int j = 0; j < 16; ++j) p += s[j] * kb[c * 16 + j];
        part[tid] = p;
        __syncthreads();
        float Sk = part[pb] + part[pb + 1] + part[pb + 2] + part[pb + 3];

        // S = a*S + be*(v_r - a*Sk_r) * k_col   (algebraic refactor of ref)
        const float coef = be * (vb[r] - a * Sk);
#pragma unroll
        for (int j = 0; j < 16; ++j) s[j] = a * s[j] + coef * kb[c * 16 + j];
        __syncthreads();

        // out = S @ q
        p = 0.f;
#pragma unroll
        for (int j = 0; j < 16; ++j) p += s[j] * qb[c * 16 + j];
        part[tid] = p;
        __syncthreads();
        if (c == 0) {
            float o = part[pb] + part[pb + 1] + part[pb + 2] + part[pb + 3];
            outs[n * D_DIM + h * 64 + r] = o;
        }
    }
#pragma unroll
    for (int j = 0; j < 16; ++j) Sfin[sbase + j] = s[j];
}

// ---------------------------------------------------------------------------
extern "C" void kernel_launch(void* const* d_in, const int* in_sizes, int n_in,
                              void* d_out, int out_size, void* d_ws, size_t ws_size,
                              hipStream_t stream) {
    const float* x  = (const float*)d_in[0];
    const float* S0 = (const float*)d_in[1];
    const float* Wq = (const float*)d_in[2];
    const float* Wk = (const float*)d_in[3];
    const float* Wv = (const float*)d_in[4];
    const float* Wa = (const float*)d_in[5];
    const float* ba = (const float*)d_in[6];
    const float* Wb = (const float*)d_in[7];
    const float* bb = (const float*)d_in[8];

    // Workspace layout (~124 MB)
    char* ws = (char*)d_ws;
    unsigned short* xbf = (unsigned short*)ws; ws += (size_t)N_TOK * D_DIM * 2;
    unsigned short* wqb = (unsigned short*)ws; ws += (size_t)D_DIM * D_DIM * 2;
    unsigned short* wkb = (unsigned short*)ws; ws += (size_t)D_DIM * D_DIM * 2;
    unsigned short* wvb = (unsigned short*)ws; ws += (size_t)D_DIM * D_DIM * 2;
    unsigned short* wab = (unsigned short*)ws; ws += (size_t)D_DIM * D_DIM * 2;
    unsigned short* wbb = (unsigned short*)ws; ws += (size_t)D_DIM * D_DIM * 2;
    float* qn = (float*)ws; ws += (size_t)N_TOK * D_DIM * 4;
    float* kn = (float*)ws; ws += (size_t)N_TOK * D_DIM * 4;
    float* vn = (float*)ws; ws += (size_t)N_TOK * D_DIM * 4;
    float* ar = (float*)ws; ws += (size_t)N_TOK * H_DIM * 4;
    float* br = (float*)ws; ws += (size_t)N_TOK * H_DIM * 4;

    // Phase 0: bf16 conversion
    const int nx = N_TOK * D_DIM;
    const int nw = D_DIM * D_DIM;
    convert_f32_bf16<<<nx / 1024, 256, 0, stream>>>(x, xbf, nx);
    convert_f32_bf16<<<nw / 1024, 256, 0, stream>>>(Wq, wqb, nw);
    convert_f32_bf16<<<nw / 1024, 256, 0, stream>>>(Wk, wkb, nw);
    convert_f32_bf16<<<nw / 1024, 256, 0, stream>>>(Wv, wvb, nw);
    convert_f32_bf16<<<nw / 1024, 256, 0, stream>>>(Wa, wab, nw);
    convert_f32_bf16<<<nw / 1024, 256, 0, stream>>>(Wb, wbb, nw);

    // Phase 1: WMMA projections with fused epilogues
    dim3 gg(N_TOK / 256, D_DIM / 64); // (32, 16)
    proj_gemm<<<gg, 256, 0, stream>>>(xbf, wqb, nullptr, qn, 1); // q, l2norm
    proj_gemm<<<gg, 256, 0, stream>>>(xbf, wkb, nullptr, kn, 1); // k, l2norm
    proj_gemm<<<gg, 256, 0, stream>>>(xbf, wvb, nullptr, vn, 0); // v, raw
    proj_gemm<<<gg, 256, 0, stream>>>(xbf, wab, ba, ar, 2);      // alpha
    proj_gemm<<<gg, 256, 0, stream>>>(xbf, wbb, bb, br, 2);      // beta

    // Phase 2: recurrent scan; outputs = outs (T*B*D) then S_final
    float* outs = (float*)d_out;
    float* Sfin = outs + (size_t)T_DIM * B_DIM * D_DIM;
    scan_kernel<<<B_DIM * H_DIM, 256, 0, stream>>>(qn, kn, vn, ar, br, S0, outs, Sfin);
}
